// MSA_31980326486555
// MI455X (gfx1250) — compile-verified, matrix operand order checked by reading, not run on hardware
//
#include <hip/hip_runtime.h>
#include <hip/hip_bf16.h>
#include <math.h>

typedef __attribute__((ext_vector_type(16))) _Float16 v16h;
typedef __attribute__((ext_vector_type(8)))  _Float16 h8;
typedef __attribute__((ext_vector_type(8)))  float    v8f;

#define NPB   16384      // tokens per batch (128*128)
#define TOK   32768      // total tokens (B * NPB)
#define CCH   256        // channels
#define NHEAD 8
#define HD    32         // head dim

// ---------------------------------------------------------------------------
// WMMA core: D(16x16 f32) = A(16x32 f16) * B(32x16 f16) + C
// ---------------------------------------------------------------------------
__device__ __forceinline__ v8f wmma16(v16h a, v16h b, v8f c) {
  return __builtin_amdgcn_wmma_f32_16x16x32_f16(
      /*neg_a=*/false, a, /*neg_b=*/false, b,
      /*c_mod=*/(short)0, c, /*reuse_a=*/false, /*reuse_b=*/false);
}

// Per-lane fragment layout (ISA 7.12.2, 16-bit 16x32 A; B mirrored):
// lane l holds "row" (row0 + (l&15)); K halves t=0..7 -> kb+t, t=8..15 -> kb+16+(t-8),
// with kb = k0 + 8*(l>>4).  Two contiguous 16-byte loads per fragment.
__device__ __forceinline__ v16h load_frag_f16(const _Float16* __restrict__ p,
                                              int ld, int row0, int k0) {
  int l = threadIdx.x & 31;
  const _Float16* base = p + (size_t)(row0 + (l & 15)) * ld + (k0 + ((l >> 4) << 3));
  union { v16h v; h8 h[2]; } u;
  u.h[0] = *(const h8*)(base);
  u.h[1] = *(const h8*)(base + 16);
  return u.v;
}

// Store a 16x16 f32 accumulator as f16 where the WMMA N index is the slow
// (ld-stride) memory dim and the M index is contiguous (one b128 store/lane).
// C/D layout: lane l holds N = l&15, M = r + 8*(l>>4) for VGPR r.
__device__ __forceinline__ void store_acc_h(_Float16* __restrict__ p, int ld,
                                            int n_base, int m_base, v8f acc) {
  int l = threadIdx.x & 31;
  _Float16* base = p + (size_t)(n_base + (l & 15)) * ld + m_base + ((l >> 4) << 3);
  h8 hv;
#pragma unroll
  for (int r = 0; r < 8; ++r) hv[r] = (_Float16)acc[r];
  *(h8*)base = hv;
}

__device__ __forceinline__ void store_acc_f32_bias(float* __restrict__ p, int ld,
                                                   int n_base, int m_base, v8f acc,
                                                   const float* __restrict__ bias) {
  int l = threadIdx.x & 31;
  int c = m_base + ((l >> 4) << 3);
  float* base = p + (size_t)(n_base + (l & 15)) * ld + c;
#pragma unroll
  for (int r = 0; r < 8; ++r) base[r] = acc[r] + bias[c + r];
}

// ---------------------------------------------------------------------------
// Streaming f32 -> f16 conversion, 8 elements/thread (b128 in/out).
// ---------------------------------------------------------------------------
__global__ void cvt_f32_to_f16(const float* __restrict__ in, _Float16* __restrict__ out,
                               int n8) {
  int i = blockIdx.x * 256 + threadIdx.x;
  if (i >= n8) return;
  const float4* p = (const float4*)in + (size_t)i * 2;
  float4 a = p[0];
  float4 b = p[1];
  h8 h;
  h[0] = (_Float16)a.x; h[1] = (_Float16)a.y; h[2] = (_Float16)a.z; h[3] = (_Float16)a.w;
  h[4] = (_Float16)b.x; h[5] = (_Float16)b.y; h[6] = (_Float16)b.z; h[7] = (_Float16)b.w;
  ((h8*)out)[i] = h;
}

// ---------------------------------------------------------------------------
// Projection GEMM, channel-major output:  Ot[b][c_out][n] = sum_c X[tok][c]*W[c_out][c]
// One wave -> 64(token) x 32(c_out) tile; M = token (A from X), N = c_out (B from W).
// ---------------------------------------------------------------------------
__global__ void gemm_proj_T(const _Float16* __restrict__ X, const _Float16* __restrict__ W,
                            _Float16* __restrict__ Ot) {
  int wave = (int)((blockIdx.x * blockDim.x + threadIdx.x) >> 5);
  int ct = wave & 7;       // channel tile [0,8)
  int tt = wave >> 3;      // token tile   [0,512)
  int m0 = tt * 64;        // global token
  int n0 = ct * 32;        // out channel
  v8f acc[4][2] = {};
#pragma unroll
  for (int k0 = 0; k0 < CCH; k0 += 32) {
    if (k0 + 32 < CCH)
      __builtin_prefetch(X + (size_t)m0 * CCH + k0 + 32, 0, 1);
    v16h b0 = load_frag_f16(W, CCH, n0, k0);
    v16h b1 = load_frag_f16(W, CCH, n0 + 16, k0);
#pragma unroll
    for (int mi = 0; mi < 4; ++mi) {
      v16h a = load_frag_f16(X, CCH, m0 + 16 * mi, k0);
      acc[mi][0] = wmma16(a, b0, acc[mi][0]);
      acc[mi][1] = wmma16(a, b1, acc[mi][1]);
    }
  }
  int b  = m0 >> 14;           // batch (64-token tile never straddles batches)
  int ml = m0 & (NPB - 1);
  _Float16* base = Ot + (size_t)b * CCH * NPB;
#pragma unroll
  for (int mi = 0; mi < 4; ++mi)
#pragma unroll
    for (int ni = 0; ni < 2; ++ni)
      store_acc_h(base, NPB, n0 + 16 * ni, ml + 16 * mi, acc[mi][ni]);
}

// ---------------------------------------------------------------------------
// Projection GEMM, token-major output:  O[tok][c_out] = sum_c X[tok][c]*W[c_out][c]
// One wave -> 32(c_out) x 64(token) tile; M = c_out (A from W), N = token (B from X).
// ---------------------------------------------------------------------------
__global__ void gemm_proj_N(const _Float16* __restrict__ W, const _Float16* __restrict__ X,
                            _Float16* __restrict__ O) {
  int wave = (int)((blockIdx.x * blockDim.x + threadIdx.x) >> 5);
  int m0 = (wave & 7) * 32;    // c_out
  int n0 = (wave >> 3) * 64;   // token
  v8f acc[2][4] = {};
#pragma unroll
  for (int k0 = 0; k0 < CCH; k0 += 32) {
    if (k0 + 32 < CCH)
      __builtin_prefetch(X + (size_t)n0 * CCH + k0 + 32, 0, 1);
    v16h a0 = load_frag_f16(W, CCH, m0, k0);
    v16h a1 = load_frag_f16(W, CCH, m0 + 16, k0);
#pragma unroll
    for (int ni = 0; ni < 4; ++ni) {
      v16h bf = load_frag_f16(X, CCH, n0 + 16 * ni, k0);
      acc[0][ni] = wmma16(a0, bf, acc[0][ni]);
      acc[1][ni] = wmma16(a1, bf, acc[1][ni]);
    }
  }
#pragma unroll
  for (int mi = 0; mi < 2; ++mi)
#pragma unroll
    for (int ni = 0; ni < 4; ++ni)
      store_acc_h(O, CCH, n0 + 16 * ni, m0 + 16 * mi, acc[mi][ni]);
}

// Final projection: f16 xo @ Wproj^T + bproj -> f32 out (token-major)
__global__ void gemm_final(const _Float16* __restrict__ W, const _Float16* __restrict__ X,
                           const float* __restrict__ bias, float* __restrict__ O) {
  int wave = (int)((blockIdx.x * blockDim.x + threadIdx.x) >> 5);
  int m0 = (wave & 7) * 32;    // c_out
  int n0 = (wave >> 3) * 64;   // token
  v8f acc[2][4] = {};
#pragma unroll
  for (int k0 = 0; k0 < CCH; k0 += 32) {
    if (k0 + 32 < CCH)
      __builtin_prefetch(X + (size_t)n0 * CCH + k0 + 32, 0, 1);
    v16h a0 = load_frag_f16(W, CCH, m0, k0);
    v16h a1 = load_frag_f16(W, CCH, m0 + 16, k0);
#pragma unroll
    for (int ni = 0; ni < 4; ++ni) {
      v16h bf = load_frag_f16(X, CCH, n0 + 16 * ni, k0);
      acc[0][ni] = wmma16(a0, bf, acc[0][ni]);
      acc[1][ni] = wmma16(a1, bf, acc[1][ni]);
    }
  }
#pragma unroll
  for (int mi = 0; mi < 2; ++mi)
#pragma unroll
    for (int ni = 0; ni < 4; ++ni)
      store_acc_f32_bias(O, CCH, n0 + 16 * ni, m0 + 16 * mi, acc[mi][ni], bias);
}

// ---------------------------------------------------------------------------
// Row sum-of-squares over n for channel-major (b,256,n) arrays -> max(||.||,eps)
// ---------------------------------------------------------------------------
__global__ void row_norm(const _Float16* __restrict__ T, float* __restrict__ norms) {
  __shared__ float red[256];
  const _Float16* row = T + (size_t)blockIdx.x * NPB;
  float s = 0.f;
  for (int i = threadIdx.x; i < NPB; i += 256) {
    float v = (float)row[i];
    s += v * v;
  }
  red[threadIdx.x] = s;
  __syncthreads();
  for (int off = 128; off > 0; off >>= 1) {
    if ((int)threadIdx.x < off) red[threadIdx.x] += red[threadIdx.x + off];
    __syncthreads();
  }
  if (threadIdx.x == 0) norms[blockIdx.x] = fmaxf(sqrtf(red[0]), 1e-12f);
}

__global__ void zero_f32(float* __restrict__ p, int n) {
  int i = blockIdx.x * 256 + threadIdx.x;
  if (i < n) p[i] = 0.f;
}

// ---------------------------------------------------------------------------
// Gram: G[br][bh][d][e] += sum_{n in slice} Kt[c0+d][n] * Qt[c0+e][n]
// One wave per (branch, bh, k-slice of 512); WMMA over K = n, atomicAdd f32.
// ---------------------------------------------------------------------------
__global__ void gram_partial(const _Float16* __restrict__ Qt,
                             const _Float16* __restrict__ K0, const _Float16* __restrict__ K1,
                             const _Float16* __restrict__ K2, const _Float16* __restrict__ K3,
                             float* __restrict__ G) {
  int ks = blockIdx.x & 31;
  int bh = (blockIdx.x >> 5) & 15;
  int br = blockIdx.x >> 9;
  const _Float16* Kt = (br == 0) ? K0 : (br == 1) ? K1 : (br == 2) ? K2 : K3;
  int b = bh >> 3, h = bh & 7;
  size_t rowoff = ((size_t)b * CCH + h * HD) * NPB;
  const _Float16* kp = Kt + rowoff;
  const _Float16* qp = Qt + rowoff;
  v8f acc[2][2] = {};
  int k0 = ks * 512;
  int kend = k0 + 512;
  for (; k0 < kend; k0 += 32) {
    v16h a0 = load_frag_f16(kp, NPB, 0, k0);
    v16h a1 = load_frag_f16(kp, NPB, 16, k0);
    v16h b0 = load_frag_f16(qp, NPB, 0, k0);
    v16h b1 = load_frag_f16(qp, NPB, 16, k0);
    acc[0][0] = wmma16(a0, b0, acc[0][0]);
    acc[0][1] = wmma16(a0, b1, acc[0][1]);
    acc[1][0] = wmma16(a1, b0, acc[1][0]);
    acc[1][1] = wmma16(a1, b1, acc[1][1]);
  }
  float* g = G + ((size_t)br * 16 + bh) * 1024;
  int l = threadIdx.x & 31;
#pragma unroll
  for (int mi = 0; mi < 2; ++mi)
#pragma unroll
    for (int ni = 0; ni < 2; ++ni)
#pragma unroll
      for (int r = 0; r < 8; ++r) {
        int d = mi * 16 + r + ((l >> 4) << 3);
        int e = ni * 16 + (l & 15);
        atomicAdd(&g[d * 32 + e], acc[mi][ni][r]);
      }
}

// ---------------------------------------------------------------------------
// Softmax over e of G/(||k_d||*||q_e||)*rescale[h]  -> attn f16 [br][bh][32][32]
// (x8 branch uses rescale4, faithful to the reference source.)
// ---------------------------------------------------------------------------
__global__ void attn_softmax(const float* __restrict__ G, const float* __restrict__ qn,
                             const float* __restrict__ kn,
                             const float* __restrict__ rm, const float* __restrict__ r2,
                             const float* __restrict__ r4,
                             _Float16* __restrict__ attnH) {
  int bh = blockIdx.x & 15;
  int br = blockIdx.x >> 4;
  int b = bh >> 3, h = bh & 7;
  const float* g   = G + (size_t)blockIdx.x * 1024;
  const float* knb = kn + (size_t)br * 512 + b * CCH + h * HD;
  const float* qnb = qn + b * CCH + h * HD;
  float rs = (br == 0 ? rm : br == 1 ? r2 : r4)[h];
  int d = threadIdx.x;  // blockDim.x == 32
  float scale_d = rs / knb[d];
  float row[32];
  float mx = -3.0e38f;
#pragma unroll
  for (int e = 0; e < 32; ++e) {
    float v = g[d * 32 + e] * scale_d / qnb[e];
    row[e] = v;
    mx = fmaxf(mx, v);
  }
  float sum = 0.f;
#pragma unroll
  for (int e = 0; e < 32; ++e) { row[e] = __expf(row[e] - mx); sum += row[e]; }
  float inv = 1.f / sum;
  _Float16* ao = attnH + (size_t)blockIdx.x * 1024 + d * 32;
#pragma unroll
  for (int e = 0; e < 32; ++e) ao[e] = (_Float16)(row[e] * inv);
}

// ---------------------------------------------------------------------------
// out[d][n] = sum_br sum_e attn[br][d][e] * V_br[b][n][h*32+e], summed over
// the 4 branches in-register; stores xo in (b,n,c) f16 for the final GEMM.
// One wave per (b, head, 16-token tile).
// ---------------------------------------------------------------------------
__global__ void attn_times_v(const _Float16* __restrict__ attnH,
                             const _Float16* __restrict__ Vm, const _Float16* __restrict__ V2,
                             const _Float16* __restrict__ V4, const _Float16* __restrict__ V8,
                             _Float16* __restrict__ xo) {
  int wave = (int)((blockIdx.x * blockDim.x + threadIdx.x) >> 5);
  int nt = wave & 1023;     // token tile of 16 within batch
  int bh = wave >> 10;
  int b = bh >> 3, h = bh & 7;
  int n0 = nt * 16;
  size_t vbase = ((size_t)b * NPB + n0) * CCH;
  v8f acc0 = {}, acc1 = {};
#pragma unroll
  for (int br = 0; br < 4; ++br) {
    const _Float16* at = attnH + ((size_t)br * 16 + bh) * 1024;
    const _Float16* V  = (br == 0) ? Vm : (br == 1) ? V2 : (br == 2) ? V4 : V8;
    v16h a0 = load_frag_f16(at, 32, 0, 0);    // d = 0..15,  K = e = 0..31
    v16h a1 = load_frag_f16(at, 32, 16, 0);   // d = 16..31
    v16h bf = load_frag_f16(V + vbase, CCH, 0, h * HD);  // B[e][token]
    acc0 = wmma16(a0, bf, acc0);
    acc1 = wmma16(a1, bf, acc1);
  }
  _Float16* xb = xo + (size_t)b * NPB * CCH;
  store_acc_h(xb, CCH, n0, h * HD + 0, acc0);
  store_acc_h(xb, CCH, n0, h * HD + 16, acc1);
}

// ---------------------------------------------------------------------------
// Positional branch: depthwise 3x3 -> exact GELU -> depthwise 3x3, added to out
// ---------------------------------------------------------------------------
__global__ void pos_conv_gelu(const _Float16* __restrict__ V, const float* __restrict__ W1,
                              _Float16* __restrict__ T) {
  int idx = blockIdx.x * 256 + threadIdx.x;
  if (idx >= 2 * NPB * CCH) return;
  int c = idx & 255;
  int x = (idx >> 8) & 127;
  int y = (idx >> 15) & 127;
  int b = idx >> 22;
  const _Float16* vb = V + (size_t)b * NPB * CCH;
  float s = 0.f;
#pragma unroll
  for (int ky = 0; ky < 3; ++ky) {
    int py = y + ky - 1;
    if (py < 0 || py > 127) continue;
#pragma unroll
    for (int kx = 0; kx < 3; ++kx) {
      int px = x + kx - 1;
      if (px < 0 || px > 127) continue;
      s += (float)vb[(size_t)(py * 128 + px) * CCH + c] * W1[c * 9 + ky * 3 + kx];
    }
  }
  float gl = 0.5f * s * (1.0f + erff(s * 0.70710678118654752f));
  T[idx] = (_Float16)gl;
}

__global__ void pos_conv_add(const _Float16* __restrict__ T, const float* __restrict__ W2,
                             float* __restrict__ Out) {
  int idx = blockIdx.x * 256 + threadIdx.x;
  if (idx >= 2 * NPB * CCH) return;
  int c = idx & 255;
  int x = (idx >> 8) & 127;
  int y = (idx >> 15) & 127;
  int b = idx >> 22;
  const _Float16* tb = T + (size_t)b * NPB * CCH;
  float s = 0.f;
#pragma unroll
  for (int ky = 0; ky < 3; ++ky) {
    int py = y + ky - 1;
    if (py < 0 || py > 127) continue;
#pragma unroll
    for (int kx = 0; kx < 3; ++kx) {
      int px = x + kx - 1;
      if (px < 0 || px > 127) continue;
      s += (float)tb[(size_t)(py * 128 + px) * CCH + c] * W2[c * 9 + ky * 3 + kx];
    }
  }
  Out[idx] += s;
}

// ---------------------------------------------------------------------------
extern "C" void kernel_launch(void* const* d_in, const int* in_sizes, int n_in,
                              void* d_out, int out_size, void* d_ws, size_t ws_size,
                              hipStream_t stream) {
  const float* x2  = (const float*)d_in[0];
  const float* x4  = (const float*)d_in[1];
  const float* x8  = (const float*)d_in[2];
  const float* y   = (const float*)d_in[3];
  const float* rsm = (const float*)d_in[13];
  const float* rs2 = (const float*)d_in[14];
  const float* rs4 = (const float*)d_in[15];
  /* d_in[16] (rescale8) intentionally unused: reference uses rescale4 for x8 */
  const float* bproj = (const float*)d_in[18];
  const float* Wp1   = (const float*)d_in[19];
  const float* Wp2   = (const float*)d_in[20];
  float* out = (float*)d_out;

  // Workspace carve-up (all regions 256B aligned)
  char* wptr = (char*)d_ws;
  auto alloc = [&](size_t bytes) -> char* {
    char* p = wptr;
    wptr += (bytes + 255) & ~(size_t)255;
    return p;
  };
  const size_t HBUF = (size_t)TOK * CCH * sizeof(_Float16);  // 16 MiB
  const size_t WELEM = (size_t)CCH * CCH;                    // 65536
  // f16 copies of the 4 activation inputs
  _Float16* hY  = (_Float16*)alloc(HBUF);
  _Float16* hX2 = (_Float16*)alloc(HBUF);
  _Float16* hX4 = (_Float16*)alloc(HBUF);
  _Float16* hX8 = (_Float16*)alloc(HBUF);
  // f16 copies of the 10 weight matrices (dict order 4..12, then Wproj)
  _Float16* hW  = (_Float16*)alloc(10 * WELEM * sizeof(_Float16));
  _Float16* hWqm = hW + 0 * WELEM, *hWkm = hW + 1 * WELEM, *hWvm = hW + 2 * WELEM;
  _Float16* hWk2 = hW + 3 * WELEM, *hWv2 = hW + 4 * WELEM;
  _Float16* hWk4 = hW + 5 * WELEM, *hWv4 = hW + 6 * WELEM;
  _Float16* hWk8 = hW + 7 * WELEM, *hWv8 = hW + 8 * WELEM;
  _Float16* hWpj = hW + 9 * WELEM;
  // intermediates
  _Float16* Qt  = (_Float16*)alloc(HBUF);   // (b,256,n) channel-major
  _Float16* Kt0 = (_Float16*)alloc(HBUF);
  _Float16* Kt1 = (_Float16*)alloc(HBUF);
  _Float16* Kt2 = (_Float16*)alloc(HBUF);
  _Float16* Kt3 = (_Float16*)alloc(HBUF);
  _Float16* Vm  = (_Float16*)alloc(HBUF);   // (b,n,256) token-major
  _Float16* V2  = (_Float16*)alloc(HBUF);
  _Float16* V4  = (_Float16*)alloc(HBUF);
  _Float16* V8  = (_Float16*)alloc(HBUF);
  _Float16* xo  = (_Float16*)alloc(HBUF);
  _Float16* Tb  = (_Float16*)alloc(HBUF);
  float* qn = (float*)alloc(512 * sizeof(float));
  float* kn = (float*)alloc(4 * 512 * sizeof(float));
  float* G  = (float*)alloc(4 * 16 * 1024 * sizeof(float));
  _Float16* attnH = (_Float16*)alloc(4 * 16 * 1024 * sizeof(_Float16));
  (void)ws_size; (void)in_sizes; (void)n_in; (void)out_size;

  // ---- precision pass: one-time f32 -> f16 conversions -------------------
  const int IN8 = TOK * CCH / 8;   // 1048576 -> 4096 blocks
  cvt_f32_to_f16<<<IN8 / 256, 256, 0, stream>>>(y,  hY,  IN8);
  cvt_f32_to_f16<<<IN8 / 256, 256, 0, stream>>>(x2, hX2, IN8);
  cvt_f32_to_f16<<<IN8 / 256, 256, 0, stream>>>(x4, hX4, IN8);
  cvt_f32_to_f16<<<IN8 / 256, 256, 0, stream>>>(x8, hX8, IN8);
  const int W8 = (int)(WELEM / 8); // 8192 -> 32 blocks
  const int widx[10] = {4, 5, 6, 7, 8, 9, 10, 11, 12, 17};
  for (int i = 0; i < 10; ++i)
    cvt_f32_to_f16<<<W8 / 256, 256, 0, stream>>>((const float*)d_in[widx[i]],
                                                 hW + (size_t)i * WELEM, W8);

  // Gram accumulator must start at zero
  zero_f32<<<(4 * 16 * 1024 + 255) / 256, 256, 0, stream>>>(G, 4 * 16 * 1024);

  const int GEMM_BLOCKS = (TOK / 64) * (CCH / 32) / 8;  // 512 blocks * 8 waves

  // Q/K projections (channel-major for WMMA-friendly Gram fragments)
  gemm_proj_T<<<GEMM_BLOCKS, 256, 0, stream>>>(hY,  hWqm, Qt);
  gemm_proj_T<<<GEMM_BLOCKS, 256, 0, stream>>>(hY,  hWkm, Kt0);
  gemm_proj_T<<<GEMM_BLOCKS, 256, 0, stream>>>(hX2, hWk2, Kt1);
  gemm_proj_T<<<GEMM_BLOCKS, 256, 0, stream>>>(hX4, hWk4, Kt2);
  gemm_proj_T<<<GEMM_BLOCKS, 256, 0, stream>>>(hX8, hWk8, Kt3);

  // V projections (token-major)
  gemm_proj_N<<<GEMM_BLOCKS, 256, 0, stream>>>(hWvm, hY,  Vm);
  gemm_proj_N<<<GEMM_BLOCKS, 256, 0, stream>>>(hWv2, hX2, V2);
  gemm_proj_N<<<GEMM_BLOCKS, 256, 0, stream>>>(hWv4, hX4, V4);
  gemm_proj_N<<<GEMM_BLOCKS, 256, 0, stream>>>(hWv8, hX8, V8);

  // L2 norms over n per (b,c) row
  row_norm<<<512, 256, 0, stream>>>(Qt,  qn);
  row_norm<<<512, 256, 0, stream>>>(Kt0, kn + 0 * 512);
  row_norm<<<512, 256, 0, stream>>>(Kt1, kn + 1 * 512);
  row_norm<<<512, 256, 0, stream>>>(Kt2, kn + 2 * 512);
  row_norm<<<512, 256, 0, stream>>>(Kt3, kn + 3 * 512);

  // Gram matrices (K split 32 ways, f32 atomics)
  gram_partial<<<4 * 16 * 32, 32, 0, stream>>>(Qt, Kt0, Kt1, Kt2, Kt3, G);

  // Normalize + rescale + softmax -> f16 attention matrices
  attn_softmax<<<64, 32, 0, stream>>>(G, qn, kn, rsm, rs2, rs4, attnH);

  // attn @ V summed over 4 branches -> xo (b,n,c) f16
  attn_times_v<<<2048, 256, 0, stream>>>(attnH, Vm, V2, V4, V8, xo);

  // Final projection + bias -> d_out f32
  gemm_final<<<GEMM_BLOCKS, 256, 0, stream>>>(hWpj, xo, bproj, out);

  // Positional branch: dwconv -> GELU -> dwconv, accumulated into d_out
  pos_conv_gelu<<<(2 * NPB * CCH + 255) / 256, 256, 0, stream>>>(Vm, Wp1, Tb);
  pos_conv_add <<<(2 * NPB * CCH + 255) / 256, 256, 0, stream>>>(Tb, Wp2, out);
}
